// Decoder_46454366273750
// MI455X (gfx1250) — matrix operand order repaired and checked by reading, not compiled, not run on hardware
//
#include <hip/hip_runtime.h>
#include <hip/hip_bf16.h>

// ---------------------------------------------------------------------------
// Problem constants (match reference setup_inputs exactly)
// ---------------------------------------------------------------------------
#define SOS   5000
#define Hh    1024
#define Zz    512
#define Cc    512
#define Vv    50000
#define NWw   512
#define GIN   2048        // H + Z + C
#define Tt    511         // n_words - 1
#define Mpad  512         // T padded to multiple of 16/64
#define G3H   3072        // 3*H

typedef __attribute__((ext_vector_type(16))) __bf16 v16bf;
typedef __attribute__((ext_vector_type(8)))  __bf16 v8bf;
typedef __attribute__((ext_vector_type(8)))  float  v8f;

__device__ __forceinline__ unsigned short f32_to_bf16(float f) {
    unsigned u = __float_as_uint(f);
    unsigned rounding = 0x7FFFu + ((u >> 16) & 1u);   // round-to-nearest-even
    return (unsigned short)((u + rounding) >> 16);
}

__device__ __forceinline__ float sigmoidf_(float x) {
    return 1.0f / (1.0f + __expf(-x));
}

// ---------------------------------------------------------------------------
// init: zero grid barrier counter, build enc = [z | context] (f32, 1024)
// ---------------------------------------------------------------------------
__global__ void k_init(const float* __restrict__ z, const float* __restrict__ ctx,
                       float* __restrict__ enc, unsigned* __restrict__ ctr) {
    int i = blockIdx.x * blockDim.x + threadIdx.x;
    if (i == 0) *ctr = 0u;
    if (i < Zz)            enc[i] = z[i];
    else if (i < Zz + Cc)  enc[i] = ctx[i - Zz];
}

// ---------------------------------------------------------------------------
// f32 -> bf16 weight conversion (grid-stride)
// ---------------------------------------------------------------------------
__global__ void k_cvt_bf16(const float* __restrict__ src,
                           unsigned short* __restrict__ dst, size_t n) {
    size_t i = (size_t)blockIdx.x * blockDim.x + threadIdx.x;
    size_t stride = (size_t)gridDim.x * blockDim.x;
    for (; i < n; i += stride) dst[i] = f32_to_bf16(src[i]);
}

// ---------------------------------------------------------------------------
// Build X (bf16, Mpad x GIN): X[t] = [embed[token_t] | enc], pad row zeroed.
// Xo gets its enc half prefilled; out half zeroed (written by recurrence).
// ---------------------------------------------------------------------------
__global__ void k_build_x(const float* __restrict__ embed, const int* __restrict__ response,
                          const float* __restrict__ enc,
                          unsigned short* __restrict__ X, unsigned short* __restrict__ Xo) {
    int idx = blockIdx.x * blockDim.x + threadIdx.x;   // Mpad * GIN threads
    int t = idx >> 11;
    int c = idx & (GIN - 1);
    if (t >= Mpad) return;
    unsigned short xe = 0, xo = 0;
    if (t < Tt) {
        if (c < Hh) {
            int tok = (t == 0) ? SOS : response[t];
            xe = f32_to_bf16(embed[(size_t)tok * Hh + c]);
        } else {
            unsigned short ev = f32_to_bf16(enc[c - Hh]);
            xe = ev;  xo = ev;
        }
    }
    X[idx] = xe;  Xo[idx] = xo;
}

// ---------------------------------------------------------------------------
// fc1: h0 = enc @ fc1_W.T + fc1_b  (1024 outputs, f32 matvec); seed h_f, h_b
// ---------------------------------------------------------------------------
__global__ void k_fc1(const float* __restrict__ enc, const float* __restrict__ W,
                      const float* __restrict__ b,
                      float* __restrict__ h_f, float* __restrict__ h_b) {
    int j = blockIdx.x * blockDim.x + threadIdx.x;
    if (j >= Hh) return;
    const float4* wr = (const float4*)(W + (size_t)j * Hh);
    const float4* ev = (const float4*)enc;
    float s = b[j];
    for (int k = 0; k < Hh / 4; ++k) {
        float4 w = wr[k], e = ev[k];
        s = fmaf(w.x, e.x, fmaf(w.y, e.y, fmaf(w.z, e.z, fmaf(w.w, e.w, s))));
    }
    h_f[j] = s;  h_b[j] = s;
}

// ---------------------------------------------------------------------------
// Generic bf16 WMMA GEMM:  out[m,n] = sum_k A[m,k]*B[n,k] + bias[n]
//   A: Mpad x K row-major bf16       B: N x K row-major bf16 (K contiguous)
//   out: f32, leading dim ldout, masked to (Mstore, Nstore)
// Block = 256 thr (8 waves). Wave tile 16M x 64N; block tile 64M x 128N.
// Fragments: 16-bit A 16x32, B 32x16 both in contiguous k = hlf*16 + s order; f32 C/D per CDNA5 ISA 7.12.2.
// K loop: 2-stage software pipeline with explicit ping-pong register buffers
// (no rotation copies); requires K % 64 == 0 (K = 2048 here).
// ---------------------------------------------------------------------------
__global__ __launch_bounds__(256)
void k_gemm_bf16(const __bf16* __restrict__ A, const __bf16* __restrict__ B,
                 const float* __restrict__ bias, float* __restrict__ out,
                 int N, int K, int Mstore, int Nstore, int ldout) {
    const int lane = threadIdx.x & 31;
    const int wave = threadIdx.x >> 5;
    const int hlf  = lane >> 4;      // 0: lanes 0-15, 1: lanes 16-31
    const int r    = lane & 15;

    const int m_tile = blockIdx.y * 64 + (wave & 3) * 16;
    const int n_base = blockIdx.x * 128 + (wave >> 2) * 64;

    v8f acc[4] = {{}, {}, {}, {}};

    const __bf16* Arow = A + (size_t)(m_tile + r) * K;

    const __bf16* Brow[4];
    int ncol[4];
#pragma unroll
    for (int j = 0; j < 4; ++j) {
        int n = n_base + j * 16 + r;
        ncol[j] = n;
        int nc = (n < N) ? n : 0;                 // clamp OOB loads; masked at store
        Brow[j] = B + (size_t)nc * K + hlf * 16;  // lane half selects K 0-15 / 16-31
    }

    union AF { v16bf v; v8bf h[2]; };

    // fragment load: A elems = 16 contiguous K halves at lane-half offset (K[k+hlf*16..])
    //                B elems = 16 contiguous K halves at lane-half offset
    auto load_frag = [&](AF& a, v16bf* b, int k) {
        a.v = *(const v16bf*)(Arow + k + hlf * 16);
#pragma unroll
        for (int j = 0; j < 4; ++j) b[j] = *(const v16bf*)(Brow[j] + k);
    };
    auto compute = [&](const AF& a, const v16bf* b) {
#pragma unroll
        for (int j = 0; j < 4; ++j)
            acc[j] = __builtin_amdgcn_wmma_f32_16x16x32_bf16(
                false, a.v, false, b[j], (short)0, acc[j], false, false);
    };

    AF a0, a1;
    v16bf b0[4], b1[4];

    // prologue
    load_frag(a0, b0, 0);

    // steady state: two stages per trip, fixed registers per stage
    int k0 = 0;
    for (; k0 + 64 < K; k0 += 64) {
        load_frag(a1, b1, k0 + 32);
        __builtin_prefetch((const void*)(Arow + k0 + 96), 0, 1);      // gfx1250
        __builtin_prefetch((const void*)(Brow[0] + k0 + 96), 0, 1);   // global_prefetch_b8
        __builtin_prefetch((const void*)(Brow[3] + k0 + 96), 0, 1);
        compute(a0, b0);
        load_frag(a0, b0, k0 + 64);
        compute(a1, b1);
    }

    // epilogue: k0 == K - 64
    load_frag(a1, b1, k0 + 32);
    compute(a0, b0);
    compute(a1, b1);

    // store D + bias (C/D layout: M = hlf*8 + vgpr index, N = lane%16)
#pragma unroll
    for (int j = 0; j < 4; ++j) {
        int n = ncol[j];
        if (n < Nstore) {
            float bv = bias[n];
#pragma unroll
            for (int rv = 0; rv < 8; ++rv) {
                int m = m_tile + hlf * 8 + rv;
                if (m < Mstore) out[(size_t)m * ldout + n] = acc[j][rv] + bv;
            }
        }
    }
}

// ---------------------------------------------------------------------------
// Persistent GRU recurrence. 48 blocks x 128 thr = 6144 threads:
//   phase 1: thread gid computes gh[gid] (f: rows 0..3071, b: rows 3072..6143)
//   phase 2: threads 0..1023 apply both gate updates for lane j, emit out_t bf16
// Two monotonic grid barriers per step.
// ---------------------------------------------------------------------------
#define RNB 48
#define RNT 128

__device__ __forceinline__ void grid_barrier(unsigned* ctr, unsigned target) {
    __syncthreads();
    if (threadIdx.x == 0) {
        __threadfence();
        __hip_atomic_fetch_add(ctr, 1u, __ATOMIC_RELEASE, __HIP_MEMORY_SCOPE_AGENT);
        while (__hip_atomic_load(ctr, __ATOMIC_ACQUIRE, __HIP_MEMORY_SCOPE_AGENT) < target)
            __builtin_amdgcn_s_sleep(2);
        __threadfence();
    }
    __syncthreads();
}

__global__ __launch_bounds__(RNT)
void k_recur(const float* __restrict__ GI_f, const float* __restrict__ GI_b,
             const float* __restrict__ Whh_f, const float* __restrict__ Whh_b,
             const float* __restrict__ bhh_f, const float* __restrict__ bhh_b,
             float* __restrict__ h_f, float* __restrict__ h_b,
             float* __restrict__ gh, unsigned short* __restrict__ Xo,
             unsigned* __restrict__ ctr) {
    const int gid  = blockIdx.x * RNT + threadIdx.x;        // 0..6143
    const bool isb = gid >= G3H;
    const int  row = isb ? gid - G3H : gid;
    const float*  W  = isb ? Whh_b : Whh_f;
    const float*  hv = isb ? h_b   : h_f;
    const float   bv = isb ? bhh_b[row] : bhh_f[row];
    const float4* wr = (const float4*)(W + (size_t)row * Hh);

    unsigned epoch = 0;
    for (int t = 0; t < Tt; ++t) {
        // ---- matvec: gh = Whh @ h + bhh (Whh resident in 192MB L2) ----
        const float4* h4 = (const float4*)hv;
        float s = bv;
        for (int k = 0; k < Hh / 4; ++k) {
            float4 w = wr[k], h = h4[k];
            s = fmaf(w.x, h.x, fmaf(w.y, h.y, fmaf(w.z, h.z, fmaf(w.w, h.w, s))));
        }
        gh[gid] = s;
        epoch += RNB;  grid_barrier(ctr, epoch);

        // ---- gates ----
        if (gid < Hh) {
            const int j = gid;
            const float* gif = GI_f + (size_t)t * G3H;
            const float* gib = GI_b + (size_t)t * G3H;

            float hf = h_f[j];
            float rf = sigmoidf_(gif[j]          + gh[j]);
            float zf = sigmoidf_(gif[Hh + j]     + gh[Hh + j]);
            float nf = tanhf    (gif[2*Hh + j]   + rf * gh[2*Hh + j]);
            hf = (1.0f - zf) * nf + zf * hf;
            h_f[j] = hf;

            float hb = h_b[j];
            float rb = sigmoidf_(gib[j]          + gh[G3H + j]);
            float zb = sigmoidf_(gib[Hh + j]     + gh[G3H + Hh + j]);
            float nb = tanhf    (gib[2*Hh + j]   + rb * gh[G3H + 2*Hh + j]);
            hb = (1.0f - zb) * nb + zb * hb;
            h_b[j] = hb;

            Xo[(size_t)t * GIN + j] = f32_to_bf16(hf + hb);  // fc2 input (out half)
        }
        epoch += RNB;  grid_barrier(ctr, epoch);
    }
}

// ---------------------------------------------------------------------------
// Host launcher
// ---------------------------------------------------------------------------
extern "C" void kernel_launch(void* const* d_in, const int* in_sizes, int n_in,
                              void* d_out, int out_size, void* d_ws, size_t ws_size,
                              hipStream_t stream) {
    const float* z        = (const float*)d_in[0];
    const float* context  = (const float*)d_in[1];
    const int*   response = (const int*)  d_in[2];
    // d_in[3] = n_words (== NWw by construction)
    const float* embed    = (const float*)d_in[4];
    const float* Wih_f    = (const float*)d_in[5];
    const float* Whh_f    = (const float*)d_in[6];
    const float* bih_f    = (const float*)d_in[7];
    const float* bhh_f    = (const float*)d_in[8];
    const float* Wih_b    = (const float*)d_in[9];
    const float* Whh_b    = (const float*)d_in[10];
    const float* bih_b    = (const float*)d_in[11];
    const float* bhh_b    = (const float*)d_in[12];
    const float* fc1_W    = (const float*)d_in[13];
    const float* fc1_b    = (const float*)d_in[14];
    const float* fc2_W    = (const float*)d_in[15];
    const float* fc2_b    = (const float*)d_in[16];
    float* out = (float*)d_out;
    (void)in_sizes; (void)n_in; (void)out_size; (void)ws_size;

    // ---- workspace carve-up (256B aligned) ----
    char* ws = (char*)d_ws;
    size_t off = 0;
    auto alloc = [&](size_t bytes) -> void* {
        void* p = ws + off;
        off += (bytes + 255) & ~(size_t)255;
        return p;
    };
    unsigned*       ctr     = (unsigned*)      alloc(256);
    float*          enc     = (float*)         alloc(1024 * sizeof(float));
    float*          h_f     = (float*)         alloc(Hh * sizeof(float));
    float*          h_b     = (float*)         alloc(Hh * sizeof(float));
    float*          gh      = (float*)         alloc(2 * G3H * sizeof(float));
    unsigned short* Wihf16  = (unsigned short*)alloc((size_t)G3H * GIN * 2);
    unsigned short* Wihb16  = (unsigned short*)alloc((size_t)G3H * GIN * 2);
    unsigned short* fc2_16  = (unsigned short*)alloc((size_t)Vv * GIN * 2);
    unsigned short* Xbf     = (unsigned short*)alloc((size_t)Mpad * GIN * 2);
    unsigned short* Xo      = (unsigned short*)alloc((size_t)Mpad * GIN * 2);
    float*          GI_f    = (float*)         alloc((size_t)Mpad * G3H * sizeof(float));
    float*          GI_b    = (float*)         alloc((size_t)Mpad * G3H * sizeof(float));

    // 1) init: barrier counter + enc
    k_init<<<4, 256, 0, stream>>>(z, context, enc, ctr);

    // 2) weight converts f32 -> bf16
    k_cvt_bf16<<<1024, 256, 0, stream>>>(Wih_f, Wihf16, (size_t)G3H * GIN);
    k_cvt_bf16<<<1024, 256, 0, stream>>>(Wih_b, Wihb16, (size_t)G3H * GIN);
    k_cvt_bf16<<<4096, 256, 0, stream>>>(fc2_W, fc2_16, (size_t)Vv * GIN);

    // 3) build GRU inputs X and fc2 input skeleton Xo
    k_build_x<<<(Mpad * GIN) / 256, 256, 0, stream>>>(embed, response, enc, Xbf, Xo);

    // 4) h0 = enc @ fc1_W.T + fc1_b
    k_fc1<<<8, 128, 0, stream>>>(enc, fc1_W, fc1_b, h_f, h_b);

    // 5) batched input projections: GI = X @ Wih.T + bih   (WMMA bf16)
    {
        dim3 grid(G3H / 128, Mpad / 64);
        k_gemm_bf16<<<grid, 256, 0, stream>>>((const __bf16*)Xbf, (const __bf16*)Wihf16,
                                              bih_f, GI_f, G3H, GIN, Mpad, G3H, G3H);
        k_gemm_bf16<<<grid, 256, 0, stream>>>((const __bf16*)Xbf, (const __bf16*)Wihb16,
                                              bih_b, GI_b, G3H, GIN, Mpad, G3H, G3H);
    }

    // 6) sequential bidirectional GRU recurrence (persistent kernel)
    k_recur<<<RNB, RNT, 0, stream>>>(GI_f, GI_b, Whh_f, Whh_b, bhh_f, bhh_b,
                                     h_f, h_b, gh, Xo, ctr);

    // 7) logits = [out | enc] @ fc2_W.T + fc2_b, keep first V-1 columns (WMMA bf16)
    {
        dim3 grid((Vv + 127) / 128, Mpad / 64);
        k_gemm_bf16<<<grid, 256, 0, stream>>>((const __bf16*)Xo, (const __bf16*)fc2_16,
                                              fc2_b, out, Vv, GIN, Tt, Vv - 1, Vv - 1);
    }
}
